// GCNEncoder_79078937854246
// MI455X (gfx1250) — matrix operand order repaired and checked by reading, not compiled, not run on hardware
//
#include <hip/hip_runtime.h>
#include <hip/hip_bf16.h>

typedef __attribute__((ext_vector_type(16))) _Float16 v16h;
typedef __attribute__((ext_vector_type(8)))  _Float16 v8h;
typedef __attribute__((ext_vector_type(4)))  _Float16 v4h;
typedef __attribute__((ext_vector_type(8)))  float    v8f;

#define D_FEAT 512

// ---------------------------------------------------------------------------
// Utility kernels
// ---------------------------------------------------------------------------

__global__ __launch_bounds__(256) void zero_f32_kernel(float* __restrict__ p, int n4) {
    int i = blockIdx.x * blockDim.x + threadIdx.x;
    if (i < n4) {
        ((float4*)p)[i] = make_float4(0.f, 0.f, 0.f, 0.f);
    }
}

__global__ __launch_bounds__(256) void f32_to_f16_kernel(const float* __restrict__ in,
                                                         _Float16* __restrict__ out, int n4) {
    int i = blockIdx.x * blockDim.x + threadIdx.x;
    if (i < n4) {
        float4 v = ((const float4*)in)[i];
        v4h h;
        h.x = (_Float16)v.x; h.y = (_Float16)v.y;
        h.z = (_Float16)v.z; h.w = (_Float16)v.w;
        ((v4h*)out)[i] = h;
    }
}

// ---------------------------------------------------------------------------
// Graph aggregation: degree count + feature scatter-add (fp32 atomics, L2-resident)
// ---------------------------------------------------------------------------

__global__ __launch_bounds__(256) void degree_kernel(const int* __restrict__ dst,
                                                     float* __restrict__ cnt, int nEdges) {
    int e = blockIdx.x * blockDim.x + threadIdx.x;
    if (e < nEdges) {
        atomicAdd(&cnt[dst[e]], 1.0f);
    }
}

// one thread per (edge, 4-float chunk): E * 128 threads
__global__ __launch_bounds__(256) void scatter_add_kernel(const float* __restrict__ x,
                                                          const int* __restrict__ src,
                                                          const int* __restrict__ dst,
                                                          float* __restrict__ sum, int nEdges) {
    int t = blockIdx.x * blockDim.x + threadIdx.x;
    int e = t >> 7;           // edge index
    int c = t & 127;          // 4-float chunk within the 512-wide feature row
    if (e < nEdges) {
        int s = src[e];
        int d = dst[e];
        float4 v = ((const float4*)(x + (size_t)s * D_FEAT))[c];
        float* o = sum + (size_t)d * D_FEAT + c * 4;
        atomicAdd(o + 0, v.x);
        atomicAdd(o + 1, v.y);
        atomicAdd(o + 2, v.z);
        atomicAdd(o + 3, v.w);
    }
}

// mean = sum / max(cnt,1), emitted directly as f16 WMMA A-operand storage
__global__ __launch_bounds__(256) void mean_to_f16_kernel(const float* __restrict__ sum,
                                                          const float* __restrict__ cnt,
                                                          _Float16* __restrict__ out, int nNodes) {
    int i = blockIdx.x * blockDim.x + threadIdx.x;   // over N*128 float4 chunks
    if (i < nNodes * (D_FEAT / 4)) {
        int node = i >> 7;
        float inv = 1.0f / fmaxf(cnt[node], 1.0f);
        float4 s = ((const float4*)sum)[i];
        v4h h;
        h.x = (_Float16)(s.x * inv); h.y = (_Float16)(s.y * inv);
        h.z = (_Float16)(s.z * inv); h.w = (_Float16)(s.w * inv);
        ((v4h*)out)[i] = h;
    }
}

// ---------------------------------------------------------------------------
// Fused dual-GEMM: out = Aagg @ Wl^T + Aself @ Wr^T + bias  (optionally ReLU)
// One wave32 per 16x16 output tile; v_wmma_f32_16x16x32_f16 over K=512.
// A and B use contiguous k order k = 16*(lane/16) + s (wave32).
// ---------------------------------------------------------------------------

template <bool RELU, bool WRITE_HALF>
__global__ __launch_bounds__(256) void sage_gemm_kernel(const _Float16* __restrict__ Aagg,
                                                        const _Float16* __restrict__ Aself,
                                                        const _Float16* __restrict__ Wl,
                                                        const _Float16* __restrict__ Wr,
                                                        const float* __restrict__ bias,
                                                        float* __restrict__ out32,
                                                        _Float16* __restrict__ outH,
                                                        int nNodes) {
    const int lane = threadIdx.x & 31;
    const int wid  = blockIdx.x * (blockDim.x >> 5) + (threadIdx.x >> 5);
    const int nTilesN = D_FEAT / 16;                  // 32
    const int mt = wid / nTilesN;
    const int nt = wid % nTilesN;
    if (mt * 16 >= nNodes) return;

    // A fragment addressing (16x32 f16): lane<16 -> K base 0, lane>=16 -> K base 16
    const int row   = mt * 16 + (lane & 15);
    const int kb    = (lane & 16) ? 16 : 0;
    // B fragment addressing (32x16 f16): column n = lane&15, K base 0 or 16
    const int col   = nt * 16 + (lane & 15);
    const int koffB = (lane & 16) ? 16 : 0;

    const _Float16* aggRow = Aagg  + (size_t)row * D_FEAT + kb;
    const _Float16* selRow = Aself + (size_t)row * D_FEAT + kb;
    const _Float16* wlRow  = Wl    + (size_t)col * D_FEAT + koffB;
    const _Float16* wrRow  = Wr    + (size_t)col * D_FEAT + koffB;

    v8f acc = {};
    for (int kk = 0; kk < D_FEAT; kk += 32) {
        union { v16h v; v8h h[2]; } aA, aS;
        aA.h[0] = *(const v8h*)(aggRow + kk);
        aA.h[1] = *(const v8h*)(aggRow + kk + 8);
        aS.h[0] = *(const v8h*)(selRow + kk);
        aS.h[1] = *(const v8h*)(selRow + kk + 8);
        v16h bL = *(const v16h*)(wlRow + kk);
        v16h bR = *(const v16h*)(wrRow + kk);
        acc = __builtin_amdgcn_wmma_f32_16x16x32_f16(false, aA.v, false, bL,
                                                     (short)0, acc, false, false);
        acc = __builtin_amdgcn_wmma_f32_16x16x32_f16(false, aS.v, false, bR,
                                                     (short)0, acc, false, false);
    }

    // Epilogue: C/D layout -> VGPR r holds (M = r + (lane<16?0:8), N = lane&15)
    const float bcol  = bias[col];
    const int   rbase = mt * 16 + ((lane & 16) ? 8 : 0);
#pragma unroll
    for (int r = 0; r < 8; ++r) {
        int   orow = rbase + r;
        float v    = acc[r] + bcol;
        if (RELU) v = v > 0.f ? v : 0.f;
        out32[(size_t)orow * D_FEAT + col] = v;
        if (WRITE_HALF) outH[(size_t)orow * D_FEAT + col] = (_Float16)v;
    }
}

// ---------------------------------------------------------------------------
// Host-side orchestration
// ---------------------------------------------------------------------------

extern "C" void kernel_launch(void* const* d_in, const int* in_sizes, int n_in,
                              void* d_out, int out_size, void* d_ws, size_t ws_size,
                              hipStream_t stream) {
    const float* x   = (const float*)d_in[0];
    const int*   ei  = (const int*)d_in[1];
    const float* Wl1 = (const float*)d_in[2];
    const float* Wr1 = (const float*)d_in[3];
    const float* b1  = (const float*)d_in[4];
    const float* Wl2 = (const float*)d_in[5];
    const float* Wr2 = (const float*)d_in[6];
    const float* b2  = (const float*)d_in[7];

    const int N = in_sizes[0] / D_FEAT;   // 50000
    const int E = in_sizes[1] / 2;        // 800000
    const int* src = ei;
    const int* dst = ei + E;

    // Workspace carve (256B aligned)
    char* p = (char*)d_ws;
    auto carve = [&](size_t bytes) -> void* {
        void* r = (void*)p;
        p += (bytes + 255) & ~(size_t)255;
        return r;
    };
    float*    sum = (float*)   carve((size_t)N * D_FEAT * 4);
    float*    cnt = (float*)   carve((size_t)N * 4);
    float*    h32 = (float*)   carve((size_t)N * D_FEAT * 4);
    _Float16* xH  = (_Float16*)carve((size_t)N * D_FEAT * 2);
    _Float16* hH  = (_Float16*)carve((size_t)N * D_FEAT * 2);
    _Float16* mH  = (_Float16*)carve((size_t)N * D_FEAT * 2);
    _Float16* w1l = (_Float16*)carve((size_t)D_FEAT * D_FEAT * 2);
    _Float16* w1r = (_Float16*)carve((size_t)D_FEAT * D_FEAT * 2);
    _Float16* w2l = (_Float16*)carve((size_t)D_FEAT * D_FEAT * 2);
    _Float16* w2r = (_Float16*)carve((size_t)D_FEAT * D_FEAT * 2);

    const int feat4   = N * (D_FEAT / 4);             // N*128 float4 chunks
    const int gFeat4  = (feat4 + 255) / 256;
    const int gCnt    = (N / 4 + 255) / 256;
    const int gEdge   = (E + 255) / 256;
    const int gScat   = ((size_t)E * 128 + 255) / 256;
    const int gW      = (D_FEAT * D_FEAT / 4 + 255) / 256;
    const int nTiles  = ((N + 15) / 16) * (D_FEAT / 16);
    const int gGemm   = (nTiles + 7) / 8;             // 8 waves per 256-thread block

    // ---- shared prep ----
    zero_f32_kernel<<<gFeat4, 256, 0, stream>>>(sum, feat4);
    zero_f32_kernel<<<gCnt, 256, 0, stream>>>(cnt, N / 4);
    degree_kernel<<<gEdge, 256, 0, stream>>>(dst, cnt, E);
    f32_to_f16_kernel<<<gW, 256, 0, stream>>>(Wl1, w1l, D_FEAT * D_FEAT / 4);
    f32_to_f16_kernel<<<gW, 256, 0, stream>>>(Wr1, w1r, D_FEAT * D_FEAT / 4);
    f32_to_f16_kernel<<<gW, 256, 0, stream>>>(Wl2, w2l, D_FEAT * D_FEAT / 4);
    f32_to_f16_kernel<<<gW, 256, 0, stream>>>(Wr2, w2r, D_FEAT * D_FEAT / 4);
    f32_to_f16_kernel<<<gFeat4, 256, 0, stream>>>(x, xH, feat4);

    // ---- layer 1 ----
    scatter_add_kernel<<<gScat, 256, 0, stream>>>(x, src, dst, sum, E);
    mean_to_f16_kernel<<<gFeat4, 256, 0, stream>>>(sum, cnt, mH, N);
    sage_gemm_kernel<true, true><<<gGemm, 256, 0, stream>>>(mH, xH, w1l, w1r, b1,
                                                            h32, hH, N);

    // ---- layer 2 ----
    zero_f32_kernel<<<gFeat4, 256, 0, stream>>>(sum, feat4);
    scatter_add_kernel<<<gScat, 256, 0, stream>>>(h32, src, dst, sum, E);
    mean_to_f16_kernel<<<gFeat4, 256, 0, stream>>>(sum, cnt, mH, N);
    sage_gemm_kernel<false, false><<<gGemm, 256, 0, stream>>>(mH, hH, w2l, w2r, b2,
                                                              (float*)d_out, nullptr, N);
}
